// HConstructor10_69363721830614
// MI455X (gfx1250) — matrix operand order repaired and checked by reading, not compiled, not run on hardware
//
#include <hip/hip_runtime.h>

typedef __attribute__((ext_vector_type(16))) __bf16 v16bf;
typedef __attribute__((ext_vector_type(8)))  __bf16 v8bf;
typedef __attribute__((ext_vector_type(4)))  __bf16 v4bf;
typedef __attribute__((ext_vector_type(8)))  float  v8f;

#define N_NODES 20000
#define DIM     512
#define T_REP   4
#define NS_EDGE 1024
#define TOTROWS (5 * N_NODES)   // (t+1)*N = 100000

#define TILE_M 128
#define TILE_N 64
#define TILE_K 32
#define LDSP   40               // LDS row pitch in halves (bank-conflict-free, 16B aligned)

// branchless ReLU on two packed bf16 in a dword: sign bit set -> zero the half
__device__ __forceinline__ unsigned relu_pk2(unsigned w) {
    unsigned s = (w >> 15) & 0x00010001u;
    return w & ~(s * 0xFFFFu);
}

// ---------------------------------------------------------------------------
// Generic WMMA GEMM:  C[z] = (reluA ? relu(A) : A) @ B[z] * alpha + bias[z]
//   A: M x K bf16 row-major.  B element (k,n) at B[z*bStride + k*sK + n*sN] (f32).
//   C row-major M x Nout (f32 or bf16).
// 128 threads = 4 waves; block tile 128x64; each wave: 32 rows x 64 cols
// (2 A-fragments x 4 B-fragments = 8 WMMA accumulators), software-pipelined
// global->reg->LDS staging.
// ---------------------------------------------------------------------------
__global__ __launch_bounds__(128)
void wmma_gemm(const __bf16* __restrict__ A, const float* __restrict__ Bm,
               const float* __restrict__ bias, void* __restrict__ Cout,
               int M, int K, int Nout,
               long bStride, long biasStride, long cStride,
               int sK, int sN, int reluA, float alpha, int outF32)
{
    __shared__ __bf16 As[TILE_M * LDSP];   // 10240 B
    __shared__ __bf16 Bs[TILE_N * LDSP];   //  5120 B

    const int tid    = threadIdx.x;
    const int lane   = tid & 31;
    const int wave   = tid >> 5;
    const int rowBlk = blockIdx.x * TILE_M;
    const int colBlk = blockIdx.y * TILE_N;
    const int z      = blockIdx.z;
    const float* B   = Bm + (long)z * bStride;

    const int laneLo = lane & 15;
    const int laneHi = (lane >> 4) & 1;
    const int kbaseA = laneHi ? 8 : 0;     // A fragment K origin (16-bit layout)
    const int kbaseB = laneHi ? 8 : 0;     // B fragment K origin

    v8f acc[2][4];
#pragma unroll
    for (int rt = 0; rt < 2; ++rt)
#pragma unroll
        for (int ct = 0; ct < 4; ++ct)
#pragma unroll
            for (int j = 0; j < 8; ++j) acc[rt][ct][j] = 0.0f;

    // staging registers (double-stage pipeline)
    uint4  aReg[4];
    float4 bReg[4];
    const int ar = tid >> 2;               // A stage: row base 0..31
    const int aq = tid & 3;                // A stage: k-octet

    auto loadGlobal = [&](int k0) {
#pragma unroll
        for (int i = 0; i < 4; ++i) {      // A: 128 rows x 32 k, 4x uint4/thread
            const int grow = rowBlk + ar + i * 32;
            if (grow < M) aReg[i] = *(const uint4*)(A + (long)grow * K + k0 + aq * 8);
            else          aReg[i] = make_uint4(0u, 0u, 0u, 0u);
        }
        if (sN == 1) {                     // row-major K x N weights: vec4 over n
#pragma unroll
            for (int i = 0; i < 4; ++i) {
                const int ch = tid + 128 * i;            // 0..511
                const int k  = ch >> 4;                   // 0..31
                const int n0 = (ch & 15) * 4;             // 0..60
                bReg[i] = *(const float4*)(B + (long)(k0 + k) * sK + (colBlk + n0));
            }
        } else {                           // sK==1 (B = HE, N x K layout): vec4 over k
#pragma unroll
            for (int i = 0; i < 4; ++i) {
                const int ch = tid + 128 * i;
                const int n  = ch >> 3;                   // 0..63
                const int kq = (ch & 7) * 4;              // 0..28
                bReg[i] = *(const float4*)(B + (long)(colBlk + n) * sN + (k0 + kq));
            }
        }
    };

    auto storeLDS = [&]() {
#pragma unroll
        for (int i = 0; i < 4; ++i) {
            uint4 w = aReg[i];
            if (reluA) {
                w.x = relu_pk2(w.x); w.y = relu_pk2(w.y);
                w.z = relu_pk2(w.z); w.w = relu_pk2(w.w);
            }
            *(uint4*)(&As[(ar + i * 32) * LDSP + aq * 8]) = w;
        }
        if (sN == 1) {
#pragma unroll
            for (int i = 0; i < 4; ++i) {
                const int ch = tid + 128 * i;
                const int k  = ch >> 4;
                const int n0 = (ch & 15) * 4;
                Bs[(n0 + 0) * LDSP + k] = (__bf16)bReg[i].x;
                Bs[(n0 + 1) * LDSP + k] = (__bf16)bReg[i].y;
                Bs[(n0 + 2) * LDSP + k] = (__bf16)bReg[i].z;
                Bs[(n0 + 3) * LDSP + k] = (__bf16)bReg[i].w;
            }
        } else {
#pragma unroll
            for (int i = 0; i < 4; ++i) {
                const int ch = tid + 128 * i;
                const int n  = ch >> 3;
                const int kq = (ch & 7) * 4;
                v4bf hv;
                hv[0] = (__bf16)bReg[i].x; hv[1] = (__bf16)bReg[i].y;
                hv[2] = (__bf16)bReg[i].z; hv[3] = (__bf16)bReg[i].w;
                *(v4bf*)(&Bs[n * LDSP + kq]) = hv;
            }
        }
    };

    auto compute = [&]() {
        const int arow0 = wave * 32 + laneLo;
        v16bf a0, a1;
        {
            v8bf lo = *(const v8bf*)&As[arow0 * LDSP + kbaseA];
            v8bf hi = *(const v8bf*)&As[arow0 * LDSP + kbaseA + 16];
#pragma unroll
            for (int j = 0; j < 8; ++j) { a0[j] = lo[j]; a0[j + 8] = hi[j]; }
        }
        {
            v8bf lo = *(const v8bf*)&As[(arow0 + 16) * LDSP + kbaseA];
            v8bf hi = *(const v8bf*)&As[(arow0 + 16) * LDSP + kbaseA + 16];
#pragma unroll
            for (int j = 0; j < 8; ++j) { a1[j] = lo[j]; a1[j + 8] = hi[j]; }
        }
#pragma unroll
        for (int ct = 0; ct < 4; ++ct) {
            const int bcol = ct * 16 + laneLo;
            v8bf lo = *(const v8bf*)&Bs[bcol * LDSP + kbaseB];
            v8bf hi = *(const v8bf*)&Bs[bcol * LDSP + kbaseB + 16];
            v16bf bfrag;
#pragma unroll
            for (int j = 0; j < 8; ++j) { bfrag[j] = lo[j]; bfrag[j + 8] = hi[j]; }
            acc[0][ct] = __builtin_amdgcn_wmma_f32_16x16x32_bf16(
                false, a0, false, bfrag, (short)0, acc[0][ct], false, false);
            acc[1][ct] = __builtin_amdgcn_wmma_f32_16x16x32_bf16(
                false, a1, false, bfrag, (short)0, acc[1][ct], false, false);
        }
    };

    // software pipeline: prefetch k+1 tile during compute of tile k
    loadGlobal(0);
    storeLDS();
    __syncthreads();
    for (int k0 = TILE_K; k0 < K; k0 += TILE_K) {
        loadGlobal(k0);
        compute();
        __syncthreads();
        storeLDS();
        __syncthreads();
    }
    compute();

    // epilogue: scale, bias, store (C layout: lane=col, vgpr v => row v + 8*laneHi)
#pragma unroll
    for (int ct = 0; ct < 4; ++ct) {
        const int col = colBlk + ct * 16 + laneLo;
        const float bv = bias ? bias[(long)z * biasStride + col] : 0.0f;
#pragma unroll
        for (int rt = 0; rt < 2; ++rt) {
#pragma unroll
            for (int v = 0; v < 8; ++v) {
                const int row = rowBlk + wave * 32 + rt * 16 + v + laneHi * 8;
                if (row < M) {
                    const float val = acc[rt][ct][v] * alpha + bv;
                    const long idx = (long)z * cStride + (long)row * Nout + col;
                    if (outF32) ((float*)Cout)[idx] = val;
                    else        ((__bf16*)Cout)[idx] = (__bf16)val;
                }
            }
        }
    }
}

// f32 -> bf16 conversion (features into all_features rows [0,N))
__global__ void cvt_bf16(const float* __restrict__ src, __bf16* __restrict__ dst, int n4)
{
    const int i = blockIdx.x * blockDim.x + threadIdx.x;
    if (i >= n4) return;
    const float4 f = *(const float4*)(src + (long)i * 4);
    v4bf h; h[0] = (__bf16)f.x; h[1] = (__bf16)f.y; h[2] = (__bf16)f.z; h[3] = (__bf16)f.w;
    *(v4bf*)(dst + (long)i * 4) = h;
}

__global__ void zero_f32(float* __restrict__ p, int n)
{
    const int i = blockIdx.x * blockDim.x + threadIdx.x;
    if (i < n) p[i] = 0.0f;
}

// one wave per row, first-occurrence argmax over 1024 logits
__global__ __launch_bounds__(256)
void argmax_rows(const float* __restrict__ logits, int* __restrict__ classes,
                 int rows, int cols)
{
    const int lane = threadIdx.x & 31;
    const int row  = blockIdx.x * 8 + (threadIdx.x >> 5);
    if (row >= rows) return;
    const float* p = logits + (long)row * cols;
    float best = -3.402823466e38f;
    int   bidx = 0x7fffffff;
    for (int c = lane; c < cols; c += 32) {
        const float v = p[c];
        if (v > best) { best = v; bidx = c; }
    }
    for (int off = 16; off > 0; off >>= 1) {
        const float ov = __shfl_down(best, off, 32);
        const int   oi = __shfl_down(bidx, off, 32);
        if (ov > best || (ov == best && oi < bidx)) { best = ov; bidx = oi; }
    }
    if (lane == 0) classes[row] = bidx;
}

// H[n, class] += 1 for each of the (t+1)*N rows  (integer-valued f32: order-safe)
__global__ void histogram(const int* __restrict__ classes, float* __restrict__ H, int rows)
{
    const int r = blockIdx.x * blockDim.x + threadIdx.x;
    if (r >= rows) return;
    const int n = r % N_NODES;
    atomicAdd(&H[(long)n * NS_EDGE + classes[r]], 1.0f);
}

// hyperedge_features[c,:] += af2[n,:] for each unique class c of node n
__global__ __launch_bounds__(256)
void he_scatter(const int* __restrict__ classes, const __bf16* __restrict__ af2,
                float* __restrict__ HE)
{
    const int n = blockIdx.x;
    int cls[5];
#pragma unroll
    for (int b = 0; b < 5; ++b) cls[b] = classes[b * N_NODES + n];
    for (int d = threadIdx.x; d < DIM; d += 256) {
        const float v = (float)af2[(long)n * DIM + d];
#pragma unroll
        for (int b = 0; b < 5; ++b) {
            bool dup = false;
#pragma unroll
            for (int j = 0; j < 5; ++j) if (j < b && cls[j] == cls[b]) dup = true;
            if (!dup) atomicAdd(&HE[(long)cls[b] * DIM + d], v);
        }
    }
}

#define ROWCHUNKS 32
__global__ void colsum_part(const float* __restrict__ H, float* __restrict__ part)
{
    const int c  = blockIdx.x * blockDim.x + threadIdx.x;     // column, coalesced
    const int ch = blockIdx.y;
    const int rp = (N_NODES + ROWCHUNKS - 1) / ROWCHUNKS;     // 625
    const int r0 = ch * rp;
    const int r1 = (r0 + rp < N_NODES) ? r0 + rp : N_NODES;
    float s = 0.0f;
    for (int n = r0; n < r1; ++n) s += __expf(H[(long)n * NS_EDGE + c]);
    part[(long)ch * NS_EDGE + c] = s;
}

__global__ void colsum_final(const float* __restrict__ part, float* __restrict__ colsum)
{
    const int c = blockIdx.x * blockDim.x + threadIdx.x;
    if (c >= NS_EDGE) return;
    float s = 0.0f;
    for (int i = 0; i < ROWCHUNKS; ++i) s += part[(long)i * NS_EDGE + c];
    colsum[c] = s;
}

__global__ void softmax_write(const float* __restrict__ H, const float* __restrict__ colsum,
                              float* __restrict__ out, int total)
{
    const int i = blockIdx.x * blockDim.x + threadIdx.x;
    if (i >= total) return;
    out[i] = __expf(H[i]) / colsum[i & (NS_EDGE - 1)];
}

// ---------------------------------------------------------------------------
extern "C" void kernel_launch(void* const* d_in, const int* in_sizes, int n_in,
                              void* d_out, int out_size, void* d_ws, size_t ws_size,
                              hipStream_t stream)
{
    const float* features = (const float*)d_in[0];
    const float* W_lin    = (const float*)d_in[1];
    const float* b_lin    = (const float*)d_in[2];
    const float* W_bb0    = (const float*)d_in[3];
    const float* b_bb0    = (const float*)d_in[4];
    const float* W_bb1    = (const float*)d_in[5];
    const float* b_bb1    = (const float*)d_in[6];
    const float* W1       = (const float*)d_in[7];
    const float* b1       = (const float*)d_in[8];

    // workspace carve-up (all 128B aligned)
    char* ws = (char*)d_ws;
    __bf16* af     = (__bf16*)ws;                       ws += (size_t)TOTROWS * DIM * 2;     // 102.4 MB
    __bf16* af2    = (__bf16*)ws;                       ws += (size_t)TOTROWS * DIM * 2;     // 102.4 MB
    char*   bigbuf = ws;                                ws += (size_t)TOTROWS * NS_EDGE * 4; // 409.6 MB (h / logits)
    float*  H      = (float*)ws;                        ws += (size_t)N_NODES * NS_EDGE * 4; // 81.9 MB
    int*    cls    = (int*)ws;                          ws += (size_t)TOTROWS * 4;
    float*  part   = (float*)ws;                        ws += (size_t)ROWCHUNKS * NS_EDGE * 4;
    float*  colsum = (float*)ws;                        ws += (size_t)NS_EDGE * 4;
    __bf16* hbuf   = (__bf16*)bigbuf;
    float*  logits = (float*)bigbuf;

    // d_out layout: H_soft (N*NS) | hyperedge_features (NS*D) | dots (5N*NS)
    float* out_Hs   = (float*)d_out;
    float* out_HE   = out_Hs + (size_t)N_NODES * NS_EDGE;
    float* out_dots = out_HE + (size_t)NS_EDGE * DIM;

    const float dotScale = 0.04419417382415922f;  // 512^-0.5
    const dim3 blk128(128), blk256(256);

    // 1. features -> bf16 rows [0,N) of all_features
    cvt_bf16<<<dim3((N_NODES * DIM / 4 + 255) / 256), blk256, 0, stream>>>(
        features, af, N_NODES * DIM / 4);

    // 2. transformed[t] = features @ W_lin[t] + b_lin[t]  -> af rows [(1+t)N, (2+t)N)
    wmma_gemm<<<dim3((N_NODES + TILE_M - 1) / TILE_M, DIM / TILE_N, T_REP), blk128, 0, stream>>>(
        af, W_lin, b_lin, af + (size_t)N_NODES * DIM,
        N_NODES, DIM, DIM,
        (long)DIM * DIM, (long)DIM, (long)N_NODES * DIM,
        DIM, 1, /*reluA=*/0, 1.0f, /*outF32=*/0);

    // 3. h = relu(all_features) @ W_bb0 + b_bb0
    wmma_gemm<<<dim3((TOTROWS + TILE_M - 1) / TILE_M, DIM / TILE_N, 1), blk128, 0, stream>>>(
        af, W_bb0, b_bb0, hbuf, TOTROWS, DIM, DIM, 0, 0, 0, DIM, 1, 1, 1.0f, 0);

    // 4. all_features2 = relu(h) @ W_bb1 + b_bb1
    wmma_gemm<<<dim3((TOTROWS + TILE_M - 1) / TILE_M, DIM / TILE_N, 1), blk128, 0, stream>>>(
        hbuf, W_bb1, b_bb1, af2, TOTROWS, DIM, DIM, 0, 0, 0, DIM, 1, 1, 1.0f, 0);

    // 5. logits = relu(all_features2) @ W1 + b1   (f32, overwrites h buffer)
    wmma_gemm<<<dim3((TOTROWS + TILE_M - 1) / TILE_M, NS_EDGE / TILE_N, 1), blk128, 0, stream>>>(
        af2, W1, b1, logits, TOTROWS, DIM, NS_EDGE, 0, 0, 0, NS_EDGE, 1, 1, 1.0f, 1);

    // 6. per-row argmax -> classes
    argmax_rows<<<dim3((TOTROWS + 7) / 8), blk256, 0, stream>>>(logits, cls, TOTROWS, NS_EDGE);

    // 7. zero H and HE output region
    zero_f32<<<dim3((N_NODES * NS_EDGE + 255) / 256), blk256, 0, stream>>>(H, N_NODES * NS_EDGE);
    zero_f32<<<dim3((NS_EDGE * DIM + 255) / 256), blk256, 0, stream>>>(out_HE, NS_EDGE * DIM);

    // 8. histogram H[n, class] += 1
    histogram<<<dim3((TOTROWS + 255) / 256), blk256, 0, stream>>>(cls, H, TOTROWS);

    // 9. hyperedge_features = (H>0).T @ all_features2[:N]  (scatter via unique classes)
    he_scatter<<<dim3(N_NODES), blk256, 0, stream>>>(cls, af2, out_HE);

    // 10. H_soft = softmax(H, axis=0): deterministic two-stage column sums of exp
    colsum_part<<<dim3(NS_EDGE / 256, ROWCHUNKS), blk256, 0, stream>>>(H, part);
    colsum_final<<<dim3((NS_EDGE + 255) / 256), blk256, 0, stream>>>(part, colsum);
    softmax_write<<<dim3((N_NODES * NS_EDGE + 255) / 256), blk256, 0, stream>>>(
        H, colsum, out_Hs, N_NODES * NS_EDGE);

    // 11. dots = all_features @ HE.T * scale   (B strides: sK=1, sN=DIM)
    wmma_gemm<<<dim3((TOTROWS + TILE_M - 1) / TILE_M, NS_EDGE / TILE_N, 1), blk128, 0, stream>>>(
        af, out_HE, nullptr, out_dots, TOTROWS, DIM, NS_EDGE, 0, 0, 0,
        1, DIM, /*reluA=*/0, dotScale, /*outF32=*/1);
}
